// MHA_34187939676793
// MI455X (gfx1250) — matrix operand order repaired and checked by reading, not compiled, not run on hardware
//
#include <hip/hip_runtime.h>

typedef unsigned short u16;
typedef unsigned int u32;
typedef __attribute__((ext_vector_type(16))) __bf16 v16bf;
typedef __attribute__((ext_vector_type(8)))  float  v8f;

union FragBF {
    v16bf v;
    uint4 q[2];
    u16   u[16];
};

__device__ inline u16 f2bf(float f) {
    u32 x = __float_as_uint(f);
    x += 0x7FFFu + ((x >> 16) & 1u);   // round-to-nearest-even
    return (u16)(x >> 16);
}

__device__ inline v8f wmma_bf16(const FragBF& a, const FragBF& b, v8f c) {
    return __builtin_amdgcn_wmma_f32_16x16x32_bf16(
        /*neg_a=*/false, a.v, /*neg_b=*/false, b.v,
        /*c_mod=*/(short)0, c, /*reuse_a=*/false, /*reuse_b=*/false);
}

// ---------------------------------------------------------------- conversions
__global__ __launch_bounds__(256) void cvt_kernel(const float* __restrict__ in,
                                                  u16* __restrict__ out, int n) {
    int i = blockIdx.x * blockDim.x + threadIdx.x;
    int stride = gridDim.x * blockDim.x;
    for (; i < n; i += stride) out[i] = f2bf(in[i]);
}

// in: [R,C] f32 row-major  ->  out: [C,R] bf16 row-major (i.e. W^T)
__global__ __launch_bounds__(256) void tpose_kernel(const float* __restrict__ in,
                                                    u16* __restrict__ out, int R, int C) {
    int idx = blockIdx.x * blockDim.x + threadIdx.x;   // over R*C, exact
    int c = idx / R;
    int r = idx - c * R;
    out[idx] = f2bf(in[(size_t)r * C + c]);
}

// ------------------------------------------------------------------ GEMM core
// C[128x64] block tile, A[M,K] bf16 row-major, BT[N,K] bf16 row-major.
// 256 threads = 8 waves; wave w computes rows [16w,16w+16) x 64 cols (4 WMMA tiles).
__device__ inline void gemm_tile_core(const u16* __restrict__ A,
                                      const u16* __restrict__ BT,
                                      int lda, int ldb, int K,
                                      int m_base, int n_base,
                                      u16* ldsA, u16* ldsB, v8f acc[4]) {
    const int tid  = threadIdx.x;
    const int lane = tid & 31;
    const int wave = tid >> 5;

    for (int k0 = 0; k0 < K; k0 += 32) {
        // ---- stage A (128x32) and B^T (64x32) into LDS, b128 per thread ----
        {
            int r = tid >> 2;            // 0..63
            int s = (tid & 3) * 8;       // 0,8,16,24 (ushorts)
            *(uint4*)(ldsA + r * 32 + s) =
                *(const uint4*)(A + (size_t)(m_base + r) * lda + k0 + s);
            *(uint4*)(ldsA + (64 + r) * 32 + s) =
                *(const uint4*)(A + (size_t)(m_base + 64 + r) * lda + k0 + s);
            *(uint4*)(ldsB + r * 32 + s) =
                *(const uint4*)(BT + (size_t)(n_base + r) * ldb + k0 + s);
            if (k0 + 32 < K)
                __builtin_prefetch(BT + (size_t)(n_base + r) * ldb + k0 + 32, 0, 3);
        }
        __syncthreads();

        // ---- per-wave WMMA: 1 A-frag, 4 B-frags, 4 wmma -------------------
        {
            const u16* la = ldsA + wave * 16 * 32;
            int row = lane & 15;
            int ka  = (lane < 16) ? 0 : 16;     // A layout: contiguous 16 K values
            FragBF a;
            a.q[0] = *(const uint4*)(la + row * 32 + ka);
            a.q[1] = *(const uint4*)(la + row * 32 + ka + 8);

            int nn = lane & 15;
            int kb = (lane < 16) ? 0 : 16;      // B layout: contiguous 16 K values, same order as A
#pragma unroll
            for (int nt = 0; nt < 4; ++nt) {
                FragBF b;
                const u16* lb = ldsB + (nt * 16 + nn) * 32 + kb;
                b.q[0] = *(const uint4*)(lb);
                b.q[1] = *(const uint4*)(lb + 8);
                acc[nt] = wmma_bf16(a, b, acc[nt]);
            }
        }
        __syncthreads();
    }
}

// ------------------------------------------------------- QKV projection GEMM
// qkv = x_bf16 @ Wqkv^T(bf16)[3072,1024] + b ; scatter into q,k [B,H,T,64], vT [B,H,64,T]
__global__ __launch_bounds__(256) void qkv_gemm_kernel(
    const u16* __restrict__ xb, const u16* __restrict__ wT,
    const float* __restrict__ bias,
    u16* __restrict__ qO, u16* __restrict__ kO, u16* __restrict__ vtO) {
    __shared__ alignas(16) u16 ldsA[128 * 32];
    __shared__ alignas(16) u16 ldsB[64 * 32];

    const int bm = blockIdx.x & 31;      // 4096/128 = 32
    const int bn = blockIdx.x >> 5;      // 3072/64  = 48
    v8f acc[4] = { {}, {}, {}, {} };
    gemm_tile_core(xb, wT, 1024, 1024, 1024, bm * 128, bn * 64, ldsA, ldsB, acc);

    const int lane  = threadIdx.x & 31;
    const int wave  = threadIdx.x >> 5;
    const int rbase = (lane < 16) ? 0 : 8;
    const int col   = lane & 15;
#pragma unroll
    for (int nt = 0; nt < 4; ++nt) {
#pragma unroll
        for (int i = 0; i < 8; ++i) {
            int m = bm * 128 + wave * 16 + rbase + i;   // global token row [0,4096)
            int c = bn * 64 + nt * 16 + col;            // [0,3072)
            u16 bv = f2bf(acc[nt][i] + bias[c]);
            int which = c >> 10;                         // 0=q 1=k 2=v
            int d  = c & 1023;
            int h  = d >> 6;
            int dk = d & 63;
            int b  = m >> 11;
            int t  = m & 2047;
            size_t bh = (size_t)(b * 16 + h);
            if (which == 0)      qO[(bh * 2048 + t) * 64 + dk] = bv;
            else if (which == 1) kO[(bh * 2048 + t) * 64 + dk] = bv;
            else                 vtO[(bh * 64 + dk) * 2048 + t] = bv;
        }
    }
}

// ------------------------------------------------------------ flash attention
// One wave per (b,h, 16-row query tile). Block = 128 threads = 4 waves.
__global__ __launch_bounds__(128) void attn_kernel(
    const u16* __restrict__ q, const u16* __restrict__ k,
    const u16* __restrict__ vt, u16* __restrict__ y) {
    __shared__ alignas(16) u16 pbuf[4][16 * 32];

    const int lane  = threadIdx.x & 31;
    const int wslot = threadIdx.x >> 5;
    const int wid   = blockIdx.x * 4 + wslot;
    const int mt = wid & 127;                 // 2048/16 tiles
    const int bh = wid >> 7;                  // b*16+h, 0..31
    const int m0 = mt * 16;

    const u16* Qb = q  + (size_t)bh * 2048 * 64;
    const u16* Kb = k  + (size_t)bh * 2048 * 64;
    const u16* Vb = vt + (size_t)bh * 64 * 2048;
    u16* pb = pbuf[wslot];

    const int col   = lane & 15;
    const int rbase = (lane < 16) ? 0 : 8;

    // Q fragments (A layout), K = 64 => two 16x32 frags
    FragBF aq[2];
    {
        int mtok = m0 + col;
        int ka = (lane < 16) ? 0 : 16;
#pragma unroll
        for (int kc = 0; kc < 2; ++kc) {
            const u16* p = Qb + mtok * 64 + kc * 32 + ka;
            aq[kc].q[0] = *(const uint4*)(p);
            aq[kc].q[1] = *(const uint4*)(p + 8);
        }
    }

    v8f acc[4] = { {}, {}, {}, {} };
    float mrun[8], lrun[8];
#pragma unroll
    for (int i = 0; i < 8; ++i) { mrun[i] = -1e30f; lrun[i] = 0.0f; }

    for (int n0 = 0; n0 < m0 + 16; n0 += 32) {
        // ---- logits 16x32 : 4 wmma (2 n-tiles x 2 K-chunks) ----
        v8f c0 = {}, c1 = {};
#pragma unroll
        for (int kc = 0; kc < 2; ++kc) {
            int kb = kc * 32 + ((lane < 16) ? 0 : 16);
            int nt0 = n0 + col;      if (nt0 > 2047) nt0 = 2047;   // clamp OOB (masked)
            int nt1 = n0 + 16 + col; if (nt1 > 2047) nt1 = 2047;
            FragBF b0, b1;
            const u16* p0 = Kb + nt0 * 64 + kb;
            b0.q[0] = *(const uint4*)(p0); b0.q[1] = *(const uint4*)(p0 + 8);
            const u16* p1 = Kb + nt1 * 64 + kb;
            b1.q[0] = *(const uint4*)(p1); b1.q[1] = *(const uint4*)(p1 + 8);
            c0 = wmma_bf16(aq[kc], b0, c0);
            c1 = wmma_bf16(aq[kc], b1, c1);
        }
        // ---- online softmax (row stats via 16-lane butterflies) ----
#pragma unroll
        for (int i = 0; i < 8; ++i) {
            int m = m0 + rbase + i;
            float l0 = c0[i] * 0.125f;                  // 1/sqrt(64)
            float l1 = c1[i] * 0.125f;
            if (n0 + col > m)      l0 = -1e30f;         // causal mask
            if (n0 + 16 + col > m) l1 = -1e30f;
            float rm = fmaxf(l0, l1);
            rm = fmaxf(rm, __shfl_xor(rm, 1, 32));
            rm = fmaxf(rm, __shfl_xor(rm, 2, 32));
            rm = fmaxf(rm, __shfl_xor(rm, 4, 32));
            rm = fmaxf(rm, __shfl_xor(rm, 8, 32));
            float mnew = fmaxf(mrun[i], rm);
            float corr = __expf(mrun[i] - mnew);
            mrun[i] = mnew;
            float p0 = __expf(l0 - mnew);
            float p1 = __expf(l1 - mnew);
            float rs = p0 + p1;
            rs += __shfl_xor(rs, 1, 32);
            rs += __shfl_xor(rs, 2, 32);
            rs += __shfl_xor(rs, 4, 32);
            rs += __shfl_xor(rs, 8, 32);
            lrun[i] = lrun[i] * corr + rs;
            acc[0][i] *= corr; acc[1][i] *= corr;
            acc[2][i] *= corr; acc[3][i] *= corr;
            pb[(rbase + i) * 32 + col]      = f2bf(p0);   // C-layout -> LDS row-major
            pb[(rbase + i) * 32 + 16 + col] = f2bf(p1);
        }
        // ---- P(16x32) @ V-chunk(32x64): A-frag from LDS, 4 wmma ----
        FragBF pa;
        {
            int ka = (lane < 16) ? 0 : 16;
            const u16* p = pb + col * 32 + ka;
            pa.q[0] = *(const uint4*)(p);
            pa.q[1] = *(const uint4*)(p + 8);
        }
        int tb = n0 + ((lane < 16) ? 0 : 16);
        if (tb + 16 > 2048) tb = 2048 - 16;              // clamp (P there is 0)
#pragma unroll
        for (int t = 0; t < 4; ++t) {
            int dv = t * 16 + col;
            const u16* p = Vb + dv * 2048 + tb;
            FragBF b;
            b.q[0] = *(const uint4*)(p); b.q[1] = *(const uint4*)(p + 8);
            acc[t] = wmma_bf16(pa, b, acc[t]);
        }
    }

    // ---- normalize + store y (bf16, [B,T,1024]) ----
    const int b = bh >> 4;
    const int h = bh & 15;
#pragma unroll
    for (int i = 0; i < 8; ++i) {
        float inv = 1.0f / lrun[i];
        int m = m0 + rbase + i;
        size_t row = (size_t)(b * 2048 + m) * 1024 + h * 64;
#pragma unroll
        for (int t = 0; t < 4; ++t)
            y[row + t * 16 + col] = f2bf(acc[t][i] * inv);
    }
}

// ------------------------------------------------------- output projection
__global__ __launch_bounds__(256) void out_gemm_kernel(
    const u16* __restrict__ yb, const u16* __restrict__ wT,
    const float* __restrict__ bias, float* __restrict__ out) {
    __shared__ alignas(16) u16 ldsA[128 * 32];
    __shared__ alignas(16) u16 ldsB[64 * 32];

    const int bm = blockIdx.x & 31;      // 4096/128
    const int bn = blockIdx.x >> 5;      // 1024/64 = 16
    v8f acc[4] = { {}, {}, {}, {} };
    gemm_tile_core(yb, wT, 1024, 1024, 1024, bm * 128, bn * 64, ldsA, ldsB, acc);

    const int lane  = threadIdx.x & 31;
    const int wave  = threadIdx.x >> 5;
    const int rbase = (lane < 16) ? 0 : 8;
    const int col   = lane & 15;
#pragma unroll
    for (int nt = 0; nt < 4; ++nt) {
#pragma unroll
        for (int i = 0; i < 8; ++i) {
            int m = bm * 128 + wave * 16 + rbase + i;
            int c = bn * 64 + nt * 16 + col;
            out[(size_t)m * 1024 + c] = acc[nt][i] + bias[c];
        }
    }
}

// ---------------------------------------------------------------- launcher
extern "C" void kernel_launch(void* const* d_in, const int* in_sizes, int n_in,
                              void* d_out, int out_size, void* d_ws, size_t ws_size,
                              hipStream_t stream) {
    (void)in_sizes; (void)n_in; (void)out_size; (void)ws_size;
    const float* x     = (const float*)d_in[0];   // [2,2048,1024]
    const float* W_qkv = (const float*)d_in[1];   // [1024,3072]
    const float* b_qkv = (const float*)d_in[2];   // [3072]
    const float* W_o   = (const float*)d_in[3];   // [1024,1024]
    const float* b_o   = (const float*)d_in[4];   // [1024]
    float* out = (float*)d_out;                   // [2,2048,1024] f32

    // workspace layout (ushort units), ~48 MB total
    u16* ws    = (u16*)d_ws;
    u16* xb    = ws;                         // 4096*1024
    u16* wqkvT = xb    + 4194304;            // 3072*1024
    u16* woT   = wqkvT + 3145728;            // 1024*1024
    u16* qb    = woT   + 1048576;            // [B,H,T,64]
    u16* kb    = qb    + 4194304;            // [B,H,T,64]
    u16* vtb   = kb    + 4194304;            // [B,H,64,T]
    u16* yb    = vtb   + 4194304;            // [B,T,1024]

    cvt_kernel  <<<2048, 256, 0, stream>>>(x, xb, 4194304);
    tpose_kernel<<<12288, 256, 0, stream>>>(W_qkv, wqkvT, 1024, 3072);
    tpose_kernel<<<4096, 256, 0, stream>>>(W_o, woT, 1024, 1024);
    qkv_gemm_kernel<<<32 * 48, 256, 0, stream>>>(xb, wqkvT, b_qkv, qb, kb, vtb);
    attn_kernel<<<1024, 128, 0, stream>>>(qb, kb, vtb, yb);
    out_gemm_kernel<<<32 * 16, 256, 0, stream>>>(yb, woT, b_o, out);
}
